// BoxFilter_11029476016223
// MI455X (gfx1250) — matrix operand order repaired and checked by reading, not compiled, not run on hardware
//
#include <hip/hip_runtime.h>

// Box filter 31x31 (truncated at edges) over channel-summed [4,3,2048,2048] f32.
// Pass 1: channel sum + vertical running-window box  -> v in d_ws (64 MiB).
// Pass 2: horizontal box as banded-ones matmul via v_wmma_f32_16x16x32_f16.
// Memory-bound: ~256 MiB HBM traffic -> ~11 us at 23.3 TB/s; v stays in 192MB L2.

typedef __attribute__((ext_vector_type(16))) _Float16 v16h;
typedef __attribute__((ext_vector_type(8)))  float    v8f;

#define RAD   15
#define HSEG  128
#define LDS_STRIDE 196   // even stride -> 16B-aligned b128 gathers per A quad

// ---------------- Pass 1: vertical box of channel sum ----------------
// grid: (W/256, H/HSEG, B), block: 256 threads, one column per thread.
__global__ void vbox_kernel(const float* __restrict__ x, float* __restrict__ v,
                            int H, int W) {
    __shared__ float ring[32][256];           // last 32 channel-summed rows
    const int tid = threadIdx.x;
    const int col = blockIdx.x * 256 + tid;
    const int h0  = blockIdx.y * HSEG;
    const int b   = blockIdx.z;
    const size_t plane = (size_t)H * (size_t)W;
    const float* x0 = x + (size_t)b * 3u * plane + (size_t)col;
    float* vp = v + (size_t)b * plane + (size_t)col;

    // Prologue: window [h0-15, h0+15] clamped; fill ring.
    float ws = 0.0f;
    for (int j = h0 - RAD; j <= h0 + RAD; ++j) {
        if (j >= 0 && j < H) {
            size_t o = (size_t)j * (size_t)W;
            float s = x0[o] + x0[plane + o] + x0[2u * plane + o];
            ring[j & 31][tid] = s;
            ws += s;
        }
    }
    const int hend = h0 + HSEG;
    for (int h = h0; h < hend; ++h) {
        vp[(size_t)h * (size_t)W] = ws;
        int jn = h + RAD + 1;                 // enters window of h+1
        if (jn < H) {
            size_t o = (size_t)jn * (size_t)W;
            float s = x0[o] + x0[plane + o] + x0[2u * plane + o];
            ring[jn & 31][tid] = s;
            ws += s;
        }
        int jo = h - RAD;                     // leaves window of h+1
        if (jo >= 0) ws -= ring[jo & 31][tid];
    }
}

// ---------------- Pass 2: horizontal box as WMMA matmul ----------------
// grid: (W/128, H/16, B), block: 256 threads = 8 waves, one 16x16 tile/wave.
// OUT(16x16) = A(16x64) * B(64x16): A = v rows over the 46-col halo (K padded
// to 64), B = banded 0/1 selector implementing the truncated 31-wide window.
// Band + image-bounds conditions collapse to ONE contiguous K-range per lane.
__global__ void hbox_wmma_kernel(const float* __restrict__ v,
                                 float* __restrict__ out, int H, int W) {
    __shared__ float tile[16 * LDS_STRIDE];
    const int tid = threadIdx.x;
    const int wb0 = blockIdx.x * 128;         // first output col of block
    const int h0  = blockIdx.y * 16;
    const int b   = blockIdx.z;
    const size_t plane = (size_t)H * (size_t)W;
    const float* vp = v + (size_t)b * plane;

    // Stage stripe: tile[r][i] = v[h0+r][wb0-15+i], zero outside image.
    for (int i = tid; i < 16 * LDS_STRIDE; i += 256) {
        int r = i / LDS_STRIDE;
        int c = i - r * LDS_STRIDE;
        int g = wb0 - 15 + c;
        float val = 0.0f;
        if ((unsigned)g < (unsigned)W)
            val = vp[(size_t)(h0 + r) * (size_t)W + (size_t)g];
        tile[i] = val;
    }
    __syncthreads();

    const int wv    = tid >> 5;               // wave id -> tile column
    const int lane  = tid & 31;
    const int w0    = wb0 + wv * 16;          // first output col of this tile
    const int n     = lane & 15;              // N index (and A row M)
    const int lhalf = lane >> 4;              // lane half selects K sub-range

    // Valid K range for output col n of this tile (band AND image bounds):
    //   lo = max(n, 15 - w0),  hi = min(n + 30, W - 1 + 15 - w0)
    int lo   = n        > (RAD - w0)         ? n        : (RAD - w0);
    int hi   = n + 2*RAD < (W - 1 + RAD - w0) ? n + 2*RAD : (W - 1 + RAD - w0);
    int span = hi - lo;                       // >= 0 (window never empty)
    int bexp = lhalf * 16 - lo;               // K = bexp-base + const per slot

    const _Float16 hONE  = (_Float16)1.0f;
    const _Float16 hZERO = (_Float16)0.0f;

    v8f acc = {};
#pragma unroll
    for (int c = 0; c < 2; ++c) {             // K chunks of 32
        v16h A, Bm;
#pragma unroll
        for (int hp = 0; hp < 8; ++hp) {      // 8 half-pairs per fragment
            int hh = 2 * hp;
            // A K order chosen here (contiguous, matches B): K = 16*lanehalf + hh
            int Ka = 32 * c + lhalf * 16 + hh;
            // global col g = w0 - 15 + Ka -> LDS col = 16*wv + Ka (even, <=175)
            int idx = n * LDS_STRIDE + (16 * wv + Ka);
            float2 q = *(const float2*)&tile[idx];   // merges into ds_load_b128
            A[hh]     = (_Float16)q.x;
            A[hh + 1] = (_Float16)q.y;

            // B K order chosen here (contiguous, matches A): K = 16*lanehalf + hh, N = lane&15.
            // Single unsigned range test: 0 <= K - lo <= span.
#pragma unroll
            for (int e = 0; e < 2; ++e) {
                int rel = bexp + (32 * c + hh + e);   // K - lo
                Bm[hh + e] = ((unsigned)rel <= (unsigned)span) ? hONE : hZERO;
            }
        }
        acc = __builtin_amdgcn_wmma_f32_16x16x32_f16(
            /*neg_a=*/false, A, /*neg_b=*/false, Bm,
            /*c_mod=*/(short)0, acc, /*reuse_a=*/false, /*reuse_b=*/false);
    }

    float* op = out + (size_t)b * plane;
#pragma unroll
    for (int j = 0; j < 8; ++j) {             // C/D: VGPR j -> M=j / M=8+j
        int row = h0 + j + lhalf * 8;
        op[(size_t)row * (size_t)W + (size_t)(w0 + n)] = acc[j];
    }
}

extern "C" void kernel_launch(void* const* d_in, const int* in_sizes, int n_in,
                              void* d_out, int out_size, void* d_ws, size_t ws_size,
                              hipStream_t stream) {
    const float* x = (const float*)d_in[0];   // [4,3,2048,2048] f32
    float* out = (float*)d_out;               // [4,1,2048,2048] f32
    float* v   = (float*)d_ws;                // intermediate, 64 MiB

    const int B = 4, H = 2048, W = 2048;
    (void)in_sizes; (void)n_in; (void)out_size; (void)ws_size;

    dim3 g1(W / 256, H / HSEG, B);            // (8, 16, 4)
    vbox_kernel<<<g1, 256, 0, stream>>>(x, v, H, W);

    dim3 g2(W / 128, H / 16, B);              // (16, 128, 4)
    hbox_wmma_kernel<<<g2, 256, 0, stream>>>(v, out, H, W);
}
